// Module_51668456570887
// MI455X (gfx1250) — compile-verified, matrix operand order checked by reading, not run on hardware
//
#include <hip/hip_runtime.h>
#include <math.h>

// Problem constants (M=96 memory slots, B=128 batch, T=32 steps, H=768)
#define M_SLOTS 96
#define BATCH   128
#define TSTEPS  32
#define HDIM    768
#define HDIM2   1536
#define ROWS_MB (M_SLOTS * BATCH) /* 12288 */

typedef __attribute__((ext_vector_type(16))) __bf16 v16bf;
typedef __attribute__((ext_vector_type(8)))  float  v8f;
typedef __attribute__((ext_vector_type(4)))  unsigned int u32x4;
typedef __attribute__((ext_vector_type(4)))  int        v4i_t;

union Frag {
    u32x4 u[2];
    v16bf v;
};

#if defined(__has_builtin)
#if __has_builtin(__builtin_amdgcn_global_load_async_to_lds_b128)
#define ASYNC_LDS 1
#endif
#endif

__device__ __forceinline__ unsigned short f2bf(float f) {
    unsigned int u = __float_as_uint(f);
    unsigned int r = (u + 0x7FFFu + ((u >> 16) & 1u)) >> 16;  // RNE
    return (unsigned short)r;
}
__device__ __forceinline__ float bf2f(unsigned short h) {
    return __uint_as_float(((unsigned int)h) << 16);
}

__device__ __forceinline__ void wait_async0() {
#if defined(__has_builtin)
#if __has_builtin(__builtin_amdgcn_s_wait_asynccnt)
    __builtin_amdgcn_s_wait_asynccnt(0);
#else
    asm volatile("s_wait_asynccnt 0x0" ::: "memory");
#endif
#else
    asm volatile("s_wait_asynccnt 0x0" ::: "memory");
#endif
}

// Stage NITER*256 b128 chunks from global to LDS. Uses the CDNA5 async
// LDS-DMA path (GLOBAL_LOAD_ASYNC_TO_LDS_B128, ASYNCcnt) when available;
// falls back to a VGPR round-trip copy otherwise.
template <int NITER>
__device__ __forceinline__ void stage_tile(const unsigned short* __restrict__ gsrc,
                                           unsigned short* __restrict__ ldst, int tid) {
#ifdef ASYNC_LDS
#pragma unroll
    for (int i = 0; i < NITER; ++i) {
        size_t e = (size_t)(tid + i * 256) * 8;  // 8 ushorts = 16 bytes
        __builtin_amdgcn_global_load_async_to_lds_b128(
            (__attribute__((address_space(1))) v4i_t*)(size_t)(gsrc + e),
            (__attribute__((address_space(3))) v4i_t*)(size_t)(ldst + e), 0, 0);
    }
    wait_async0();
#else
    const u32x4* src = (const u32x4*)gsrc;
    u32x4* dst = (u32x4*)ldst;
#pragma unroll
    for (int i = 0; i < NITER; ++i) dst[tid + i * 256] = src[tid + i * 256];
#endif
}

// ---------------------------------------------------------------------------
// Pack a [HDIM x HDIM] fp32 weight W (out-major: W[o][h]) into a contiguous-k WMMA
// B-fragment layout for D = A(16x32 bf16) * B(32x16 bf16):
//   packed[((kt*48 + nt)*32 + lane)*16 + e]
//     = bf16(W[nt*16 + (lane&15)][kt*32 + (lane>=16?16:0) + e])
// so a wave's B fragment is 32 contiguous bytes per lane.
// ---------------------------------------------------------------------------
__global__ void k_pack_w(const float* __restrict__ W, unsigned short* __restrict__ P) {
    int tid  = blockIdx.x * blockDim.x + threadIdx.x;  // 36864 threads total
    int lane = tid & 31;
    int tile = tid >> 5;            // 0..1151
    int nt   = tile % 48;
    int kt   = tile / 48;
    int o    = nt * 16 + (lane & 15);
    int kb   = kt * 32 + ((lane >> 4) & 1) * 16;
    unsigned short* dst = P + (size_t)tid * 16;
#pragma unroll
    for (int e = 0; e < 16; ++e)
        dst[e] = f2bf(W[(size_t)o * HDIM + kb + e]);
}

// ---------------------------------------------------------------------------
// fp32 -> bf16 conversions + recurrent-state init (value = eff = mem_key)
// ---------------------------------------------------------------------------
__global__ void k_cvt_init(const float* __restrict__ mk, const float* __restrict__ inp,
                           unsigned short* __restrict__ mkb, unsigned short* __restrict__ val,
                           unsigned short* __restrict__ eff, unsigned short* __restrict__ inpb) {
    size_t i = (size_t)blockIdx.x * 256 + threadIdx.x;
    if (i < (size_t)ROWS_MB * HDIM) {
        unsigned short b = f2bf(mk[i]);
        mkb[i] = b; val[i] = b; eff[i] = b;
    }
    if (i < (size_t)TSTEPS * BATCH * HDIM)
        inpb[i] = f2bf(inp[i]);
}

// ---------------------------------------------------------------------------
// 12-tile WMMA sweep for one 32-wide K slab: base-pointer + immediate-offset
// B loads (tiles are j*1024 bytes apart), double-buffered so load j+1
// overlaps WMMA j.
// ---------------------------------------------------------------------------
__device__ __forceinline__ void gemm_kslab(const Frag& a, const char* bptr, v8f acc[12]) {
    const u32x4* bp = (const u32x4*)bptr;
    Frag bb[2];
    bb[0].u[0] = bp[0];
    bb[0].u[1] = bp[1];
#pragma unroll
    for (int j = 0; j < 12; ++j) {
        if (j < 11) {
            bb[(j + 1) & 1].u[0] = bp[(j + 1) * 64 + 0];
            bb[(j + 1) & 1].u[1] = bp[(j + 1) * 64 + 1];
        }
        acc[j] = __builtin_amdgcn_wmma_f32_16x16x32_bf16(
            false, a.v, false, bb[j & 1].v, (short)0, acc[j], false, false);
    }
}

// ---------------------------------------------------------------------------
// Generic projection GEMM: Out[r][o] = bf16( sum_h A[r][h]*W[o][h] + bias[o] )
// A: [rows x 768] bf16 row-major, BP: packed weight, per-WG tile 32x768.
// ---------------------------------------------------------------------------
__global__ __launch_bounds__(256) void k_gemm_proj(
        const unsigned short* __restrict__ A, const unsigned short* __restrict__ BP,
        const float* __restrict__ bias, unsigned short* __restrict__ Out) {
    __shared__ unsigned short sA[32 * HDIM];  // 48 KB
    const int tid  = threadIdx.x;
    const int wave = tid >> 5, lane = tid & 31;
    const int rowBase = blockIdx.x * 32;

    stage_tile<12>(A + (size_t)rowBase * HDIM, sA, tid);
    __syncthreads();

    const int sub  = wave & 1;        // row sub-tile (16 rows each)
    const int cg   = wave >> 1;       // column group (192 cols each)
    const int hi   = (lane >> 4) & 1;
    const int lrow = sub * 16 + (lane & 15);

    v8f acc[12];
    const v8f vzero = {0.f, 0.f, 0.f, 0.f, 0.f, 0.f, 0.f, 0.f};
#pragma unroll
    for (int j = 0; j < 12; ++j) acc[j] = vzero;

    const char* bptr = (const char*)BP + (size_t)(cg * 12 * 32 + lane) * 32;
    for (int kt = 0; kt < 24; ++kt) {
        Frag a;
        const u32x4* ap = (const u32x4*)(sA + lrow * HDIM + kt * 32 + hi * 16);
        a.u[0] = ap[0]; a.u[1] = ap[1];
        if (kt + 1 < 24) __builtin_prefetch(bptr + 48 * 1024, 0, 1);
        gemm_kslab(a, bptr, acc);
        bptr += 48 * 1024;
    }
#pragma unroll
    for (int j = 0; j < 12; ++j) {
        int col = cg * 192 + j * 16 + (lane & 15);
        float bc = bias[col];
#pragma unroll
        for (int e = 0; e < 8; ++e) {
            int r = rowBase + sub * 16 + hi * 8 + e;  // C layout: M = e (+8 for hi half)
            Out[(size_t)r * HDIM + col] = f2bf(acc[j][e] + bc);
        }
    }
}

// ---------------------------------------------------------------------------
// gate[m,b] = sigmoid( x_t . (mem_key[m,b] + value[m,b]) ), one wave per (m,b)
// ---------------------------------------------------------------------------
__global__ void k_gate(const float* __restrict__ mk, const float* __restrict__ inp,
                       const unsigned short* __restrict__ val,
                       float* __restrict__ gate, int t) {
    int wave = (blockIdx.x * blockDim.x + threadIdx.x) >> 5;
    int lane = threadIdx.x & 31;
    if (wave >= ROWS_MB) return;
    int b = wave & (BATCH - 1);
    const float* x  = inp + ((size_t)t * BATCH + b) * HDIM;
    const float* mr = mk + (size_t)wave * HDIM;
    const unsigned short* vr = val + (size_t)wave * HDIM;
    float g = 0.f;
    for (int h = lane; h < HDIM; h += 32)
        g += x[h] * (mr[h] + bf2f(vr[h]));
#pragma unroll
    for (int m = 16; m > 0; m >>= 1) g += __shfl_xor(g, m, 32);
    if (lane == 0) gate[wave] = 1.f / (1.f + expf(-g));
}

// ---------------------------------------------------------------------------
// One scan step: tmp = value @ Wv^T (WMMA), then
//   v = gate * relu(inp_proj + key_proj + tmp + bv) + value
//   v /= (||v||_2 + 1e-8);  value <- v;  eff <- v where input_len-1 == t
// Per-WG tile: 32 rows x 768 cols. Row-norm via LDS reduction.
// ---------------------------------------------------------------------------
__global__ __launch_bounds__(256) void k_step(
        const unsigned short* __restrict__ WvP,
        const unsigned short* __restrict__ kproj,
        const unsigned short* __restrict__ iproj,
        const float* __restrict__ bv, const float* __restrict__ gate,
        const int* __restrict__ ilen,
        unsigned short* __restrict__ val, unsigned short* __restrict__ eff, int t) {
    __shared__ unsigned short sA[32 * HDIM];  // old value tile (also the residual)
    __shared__ float sSum[32];
    __shared__ float sGate[32];
    __shared__ int   sIL[32];
    const int tid  = threadIdx.x;
    const int wave = tid >> 5, lane = tid & 31;
    const int rowBase = blockIdx.x * 32;

    stage_tile<12>(val + (size_t)rowBase * HDIM, sA, tid);
    if (tid < 32) {
        sSum[tid]  = 0.f;
        sGate[tid] = gate[rowBase + tid];
        sIL[tid]   = ilen[(rowBase + tid) & (BATCH - 1)];
    }
    __syncthreads();

    const int sub  = wave & 1;
    const int cg   = wave >> 1;
    const int hi   = (lane >> 4) & 1;
    const int lrow = sub * 16 + (lane & 15);

    v8f acc[12];
    const v8f vzero = {0.f, 0.f, 0.f, 0.f, 0.f, 0.f, 0.f, 0.f};
#pragma unroll
    for (int j = 0; j < 12; ++j) acc[j] = vzero;

    const char* bptr = (const char*)WvP + (size_t)(cg * 12 * 32 + lane) * 32;
    for (int kt = 0; kt < 24; ++kt) {
        Frag a;
        const u32x4* ap = (const u32x4*)(sA + lrow * HDIM + kt * 32 + hi * 16);
        a.u[0] = ap[0]; a.u[1] = ap[1];
        if (kt + 1 < 24) __builtin_prefetch(bptr + 48 * 1024, 0, 1);
        gemm_kslab(a, bptr, acc);
        bptr += 48 * 1024;
    }

    // Epilogue: v = gate*relu(xp + kp + acc + bv) + value_old, per-row sumsq.
    float rowsq[8];
#pragma unroll
    for (int e = 0; e < 8; ++e) rowsq[e] = 0.f;
#pragma unroll
    for (int j = 0; j < 12; ++j) {
        int col = cg * 192 + j * 16 + (lane & 15);
        float bvc = bv[col];
#pragma unroll
        for (int e = 0; e < 8; ++e) {
            int lr  = sub * 16 + hi * 8 + e;
            int row = rowBase + lr;
            int b   = row & (BATCH - 1);
            float u = acc[j][e]
                    + bf2f(kproj[(size_t)row * HDIM + col])
                    + bf2f(iproj[((size_t)t * BATCH + b) * HDIM + col])
                    + bvc;
            float r    = fmaxf(u, 0.f);
            float vold = bf2f(sA[lr * HDIM + col]);
            float vn   = sGate[lr] * r + vold;
            acc[j][e]  = vn;
            rowsq[e]  += vn * vn;
        }
    }
    // Reduce within each 16-lane half (all lanes of a half share the same 8 rows).
#pragma unroll
    for (int e = 0; e < 8; ++e) {
        float s = rowsq[e];
#pragma unroll
        for (int m = 8; m > 0; m >>= 1) s += __shfl_xor(s, m, 32);
        rowsq[e] = s;
    }
    if ((lane & 15) == 0) {
#pragma unroll
        for (int e = 0; e < 8; ++e)
            atomicAdd(&sSum[sub * 16 + hi * 8 + e], rowsq[e]);
    }
    __syncthreads();

#pragma unroll
    for (int j = 0; j < 12; ++j) {
        int col = cg * 192 + j * 16 + (lane & 15);
#pragma unroll
        for (int e = 0; e < 8; ++e) {
            int lr  = sub * 16 + hi * 8 + e;
            int row = rowBase + lr;
            float scale = 1.f / (sqrtf(sSum[lr]) + 1e-8f);
            unsigned short vb = f2bf(acc[j][e] * scale);
            val[(size_t)row * HDIM + col] = vb;
            if (sIL[lr] - 1 == t) eff[(size_t)row * HDIM + col] = vb;
        }
    }
}

// ---------------------------------------------------------------------------
// Per-(m,b) heads over kv = [eff ; mem_key] (length 1536):
//   logit (masked -> ws + identify_scores output) and entail_scores (3 cols).
// One wave per (m,b).
// ---------------------------------------------------------------------------
__global__ void k_heads(const unsigned short* __restrict__ eff, const float* __restrict__ mk,
                        const float* __restrict__ w_attn, const float* __restrict__ b_attn,
                        const float* __restrict__ w_ent,  const float* __restrict__ b_ent,
                        const unsigned char* __restrict__ mask,
                        float* __restrict__ logits, float* __restrict__ out) {
    int wave = (blockIdx.x * blockDim.x + threadIdx.x) >> 5;
    int lane = threadIdx.x & 31;
    if (wave >= ROWS_MB) return;
    const unsigned short* er = eff + (size_t)wave * HDIM;
    const float* kr = mk + (size_t)wave * HDIM;
    float a0 = 0.f, e0 = 0.f, e1 = 0.f, e2 = 0.f;
    for (int o = lane; o < HDIM; o += 32) {
        float v1 = bf2f(er[o]);
        float v2 = kr[o];
        a0 += v1 * w_attn[o]        + v2 * w_attn[HDIM + o];
        e0 += v1 * w_ent[o * 3 + 0] + v2 * w_ent[(HDIM + o) * 3 + 0];
        e1 += v1 * w_ent[o * 3 + 1] + v2 * w_ent[(HDIM + o) * 3 + 1];
        e2 += v1 * w_ent[o * 3 + 2] + v2 * w_ent[(HDIM + o) * 3 + 2];
    }
#pragma unroll
    for (int m = 16; m > 0; m >>= 1) {
        a0 += __shfl_xor(a0, m, 32);
        e0 += __shfl_xor(e0, m, 32);
        e1 += __shfl_xor(e1, m, 32);
        e2 += __shfl_xor(e2, m, 32);
    }
    if (lane == 0) {
        int mIdx = wave / BATCH, b = wave & (BATCH - 1);
        float lg = a0 + b_attn[0];
        float lm = mask[wave] ? lg : -__builtin_inff();
        logits[wave] = lm;
        out[512 + b * M_SLOTS + mIdx] = lm;                           // identify [B,M]
        float* ent = out + 512 + BATCH * M_SLOTS + (size_t)wave * 3;  // entail [M,B,3]
        ent[0] = e0 + b_ent[0];
        ent[1] = e1 + b_ent[1];
        ent[2] = e2 + b_ent[2];
    }
}

// ---------------------------------------------------------------------------
// Per-batch softmax over memory dim, decision = sum_m alpha*kv, clf = dec@w_out
// One block (256 thr) per batch element.
// ---------------------------------------------------------------------------
__global__ __launch_bounds__(256) void k_decide(
        const float* __restrict__ logits, const unsigned short* __restrict__ eff,
        const float* __restrict__ mk, const float* __restrict__ w_out,
        const float* __restrict__ b_out, float* __restrict__ out) {
    __shared__ float sAl[M_SLOTS];
    __shared__ float sDec[HDIM2];
    __shared__ float sInv;
    int b = blockIdx.x, tid = threadIdx.x;
    if (tid < M_SLOTS) sAl[tid] = logits[tid * BATCH + b];
    __syncthreads();
    if (tid == 0) {
        float mx = -__builtin_inff();
        for (int m = 0; m < M_SLOTS; ++m) mx = fmaxf(mx, sAl[m]);
        float s = 0.f;
        for (int m = 0; m < M_SLOTS; ++m) { float e = expf(sAl[m] - mx); sAl[m] = e; s += e; }
        sInv = 1.f / s;
    }
    __syncthreads();
    float inv = sInv;
    for (int o = tid; o < HDIM2; o += 256) {
        float s = 0.f;
        if (o < HDIM) {
            for (int m = 0; m < M_SLOTS; ++m)
                s += sAl[m] * bf2f(eff[((size_t)m * BATCH + b) * HDIM + o]);
        } else {
            int oo = o - HDIM;
            for (int m = 0; m < M_SLOTS; ++m)
                s += sAl[m] * mk[((size_t)m * BATCH + b) * HDIM + oo];
        }
        sDec[o] = s * inv;
    }
    __syncthreads();
    int wave = tid >> 5, lane = tid & 31;
    if (wave < 4) {
        float s = 0.f;
        for (int o = lane; o < HDIM2; o += 32) s += sDec[o] * w_out[o * 4 + wave];
#pragma unroll
        for (int m = 16; m > 0; m >>= 1) s += __shfl_xor(s, m, 32);
        if (lane == 0) out[b * 4 + wave] = s + b_out[wave];
    }
}

// ---------------------------------------------------------------------------
extern "C" void kernel_launch(void* const* d_in, const int* in_sizes, int n_in,
                              void* d_out, int out_size, void* d_ws, size_t ws_size,
                              hipStream_t stream) {
    const float* mem_key = (const float*)d_in[0];
    const float* inp     = (const float*)d_in[1];
    const int*   ilen    = (const int*)d_in[2];
    const unsigned char* mask = (const unsigned char*)d_in[3];
    const float* Wi = (const float*)d_in[4];
    const float* bi = (const float*)d_in[5];
    const float* Wk = (const float*)d_in[6];
    const float* bk = (const float*)d_in[7];
    const float* Wv = (const float*)d_in[8];
    const float* bv = (const float*)d_in[9];
    const float* w_attn = (const float*)d_in[10];
    const float* b_attn = (const float*)d_in[11];
    const float* w_out  = (const float*)d_in[12];
    const float* b_out  = (const float*)d_in[13];
    const float* w_ent  = (const float*)d_in[14];
    const float* b_ent  = (const float*)d_in[15];
    float* out = (float*)d_out;

    char* ws = (char*)d_ws;
    size_t off = 0;
    auto alloc = [&](size_t bytes) -> char* {
        char* p = ws + off;
        off += (bytes + 255) & ~(size_t)255;
        return p;
    };
    const size_t packE = 24 * 48 * 32 * 16;  // 589824 bf16 per packed weight
    unsigned short* WiP  = (unsigned short*)alloc(packE * 2);
    unsigned short* WkP  = (unsigned short*)alloc(packE * 2);
    unsigned short* WvP  = (unsigned short*)alloc(packE * 2);
    unsigned short* mkb  = (unsigned short*)alloc((size_t)ROWS_MB * HDIM * 2);
    unsigned short* valb = (unsigned short*)alloc((size_t)ROWS_MB * HDIM * 2);
    unsigned short* effb = (unsigned short*)alloc((size_t)ROWS_MB * HDIM * 2);
    unsigned short* inpb = (unsigned short*)alloc((size_t)TSTEPS * BATCH * HDIM * 2);
    unsigned short* kproj = (unsigned short*)alloc((size_t)ROWS_MB * HDIM * 2);
    unsigned short* iproj = (unsigned short*)alloc((size_t)TSTEPS * BATCH * HDIM * 2);
    float* gatep  = (float*)alloc((size_t)ROWS_MB * 4);
    float* logitp = (float*)alloc((size_t)ROWS_MB * 4);

    // Prologue: pack weights, convert inputs, init recurrent state.
    k_pack_w<<<144, 256, 0, stream>>>(Wi, WiP);
    k_pack_w<<<144, 256, 0, stream>>>(Wk, WkP);
    k_pack_w<<<144, 256, 0, stream>>>(Wv, WvP);
    k_cvt_init<<<36864, 256, 0, stream>>>(mem_key, inp, mkb, valb, effb, inpb);

    // Step-invariant projections.
    k_gemm_proj<<<ROWS_MB / 32, 256, 0, stream>>>(mkb, WkP, bk, kproj);
    k_gemm_proj<<<(TSTEPS * BATCH) / 32, 256, 0, stream>>>(inpb, WiP, bi, iproj);

    // Sequential scan: gate then fused GEMM+update per step.
    for (int t = 0; t < TSTEPS; ++t) {
        k_gate<<<ROWS_MB / 8, 256, 0, stream>>>(mem_key, inp, valb, gatep, t);
        k_step<<<ROWS_MB / 32, 256, 0, stream>>>(WvP, kproj, iproj, bv, gatep, ilen,
                                                 valb, effb, t);
    }

    // Heads: logits/identify/entail, then softmax/decision/clf.
    k_heads<<<ROWS_MB / 8, 256, 0, stream>>>(effb, mem_key, w_attn, b_attn,
                                             w_ent, b_ent, mask, logitp, out);
    k_decide<<<BATCH, 256, 0, stream>>>(logitp, effb, mem_key, w_out, b_out, out);
}
